// ClipLoss_26130581029503
// MI455X (gfx1250) — matrix operand order repaired and checked by reading, not compiled, not run on hardware
//
#include <hip/hip_runtime.h>
#include <hip/hip_bf16.h>

typedef __attribute__((ext_vector_type(16))) _Float16 v16h;
typedef __attribute__((ext_vector_type(8)))  _Float16 v8h;
typedef __attribute__((ext_vector_type(8)))  float    v8f;

#define TOPK 10
#define DMAX 1024   // feature dim of this problem (setup: D=1024); LDS sized for it

// ---------------------------------------------------------------------------
// Fragment loaders per CDNA5 ISA 7.12.2 layouts (16-bit A 16x32, B 32x16).
// A: lanes 0-15 rows M=0..15; halves 0..7 -> K = k0+8*hi.., halves 8..15 ->
// K = k0+8*hi+16.. . B (stored as row of Y = column of B): same split,
// halves 0..7 at K = k0 + 8*hi, halves 8..15 at K = k0 + 8*hi + 16.
// ---------------------------------------------------------------------------
__device__ inline v16h load_frag_a(const _Float16* p, int hi, int k0) {
    const v8h lo = *(const v8h*)(p + k0 + hi * 8);
    const v8h hh = *(const v8h*)(p + k0 + hi * 8 + 16);
    v16h r;
#pragma unroll
    for (int h = 0; h < 8; ++h) { r[h] = lo[h]; r[h + 8] = hh[h]; }
    return r;
}

__device__ inline v16h load_frag_b(const _Float16* __restrict__ p, int hi, int k0) {
    const v8h lo = *(const v8h*)(p + k0 + hi * 8);
    const v8h hh = *(const v8h*)(p + k0 + hi * 8 + 16);
    v16h r;
#pragma unroll
    for (int h = 0; h < 8; ++h) { r[h] = lo[h]; r[h + 8] = hh[h]; }
    return r;
}

// 16x16 tile: A fragments come from the LDS-staged row strip (ds_load_b128),
// B fragments stream from the L2-resident f16 matrix (global_load_b128).
__device__ inline v8f tile_gemm_lds(const _Float16* pA_lds,
                                    const _Float16* __restrict__ pB,
                                    int D, int hi) {
    v8f c = {};
    for (int k0 = 0; k0 < D; k0 += 32) {
        v16h a = load_frag_a(pA_lds, hi, k0);
        v16h b = load_frag_b(pB, hi, k0);
        c = __builtin_amdgcn_wmma_f32_16x16x32_f16(false, a, false, b,
                                                   (short)0, c, false, false);
    }
    return c;
}

// ---------------------------------------------------------------------------
// Kernel 1: normalize text rows and produce f16 copies of all three matrices.
// f16 matrices (48MB total) stay L2-resident on MI455X (192MB L2).
// ---------------------------------------------------------------------------
__global__ void prep_kernel(const float* __restrict__ img,
                            const float* __restrict__ txt,
                            _Float16* __restrict__ img_h,
                            _Float16* __restrict__ txt_h,
                            _Float16* __restrict__ tn_h,
                            int N, int D) {
    const int i = blockIdx.x;
    const size_t base = (size_t)i * D;
    float ss = 0.f;
    for (int d = threadIdx.x; d < D; d += blockDim.x) {
        float t = txt[base + d];
        ss += t * t;
    }
    __shared__ float red[8];
#pragma unroll
    for (int off = 16; off > 0; off >>= 1) ss += __shfl_down(ss, off, 32);
    const int lane = threadIdx.x & 31, w = threadIdx.x >> 5;
    if (lane == 0) red[w] = ss;
    __syncthreads();
    if (threadIdx.x == 0) {
        float t = 0.f;
        for (int k = 0; k < 8; ++k) t += red[k];
        red[0] = t;
    }
    __syncthreads();
    const float inv = 1.0f / fmaxf(sqrtf(red[0]), 1e-8f);
    for (int d = threadIdx.x; d < D; d += blockDim.x) {
        float t = txt[base + d];
        txt_h[base + d] = (_Float16)t;
        tn_h[base + d]  = (_Float16)(t * inv);
        img_h[base + d] = (_Float16)img[base + d];
    }
}

// ---------------------------------------------------------------------------
// Kernel 2: streaming row-logsumexp of scale * (X @ Y^T).
// Workgroup = 8 waves owning a 16-row strip; the 32KB A strip is staged in
// LDS once per workgroup (all waves share it); waves split the 512 column
// tiles. Per-lane online (max,sumexp), merged via shfl_xor then LDS.
// ---------------------------------------------------------------------------
__global__ void gemm_lse_kernel(const _Float16* __restrict__ X,
                                const _Float16* __restrict__ Y,
                                const float* __restrict__ scale_p,
                                float* __restrict__ outLSE,
                                int N, int D) {
    const int lane = threadIdx.x & 31;
    const int wave = threadIdx.x >> 5;
    const int hi = lane >> 4;
    const int row_base = blockIdx.x * 16;
    const float scale = *scale_p;

    __shared__ __align__(16) _Float16 As[16 * DMAX];
    {   // cooperative copy of the 16-row A strip into LDS (once per block)
        const v8h* src = (const v8h*)(X + (size_t)row_base * D);
        v8h* dst = (v8h*)As;
        const int cnt = 16 * D / 8;
        for (int idx = threadIdx.x; idx < cnt; idx += blockDim.x)
            dst[idx] = src[idx];
    }
    __syncthreads();

    const _Float16* pA = As + (lane & 15) * D;

    float m[8], s[8];
#pragma unroll
    for (int v = 0; v < 8; ++v) { m[v] = -3.0e38f; s[v] = 0.f; }

    const int ntile = N / 16;
    for (int ct = wave; ct < ntile; ct += 8) {
        const _Float16* pB = Y + (size_t)(ct * 16 + (lane & 15)) * D;
        v8f c = tile_gemm_lds(pA, pB, D, hi);
#pragma unroll
        for (int v = 0; v < 8; ++v) {
            float x = scale * c[v];
            if (x > m[v]) { s[v] = s[v] * __expf(m[v] - x) + 1.0f; m[v] = x; }
            else           s[v] += __expf(x - m[v]);
        }
    }
    // merge (m,s) across the 16 lanes of each half (same rows, diff columns)
#pragma unroll
    for (int off = 1; off < 16; off <<= 1) {
#pragma unroll
        for (int v = 0; v < 8; ++v) {
            float om = __shfl_xor(m[v], off, 32);
            float os = __shfl_xor(s[v], off, 32);
            float M = fmaxf(m[v], om);
            s[v] = s[v] * __expf(m[v] - M) + os * __expf(om - M);
            m[v] = M;
        }
    }
    __shared__ float lm[8][16];
    __shared__ float ls[8][16];
    if ((lane & 15) == 0) {
#pragma unroll
        for (int v = 0; v < 8; ++v) {
            lm[wave][v + 8 * hi] = m[v];
            ls[wave][v + 8 * hi] = s[v];
        }
    }
    __syncthreads();
    if (threadIdx.x < 16) {
        const int r = threadIdx.x;
        float M = lm[0][r];
        for (int w = 1; w < 8; ++w) M = fmaxf(M, lm[w][r]);
        float S = 0.f;
        for (int w = 0; w < 8; ++w) S += ls[w][r] * __expf(lm[w][r] - M);
        outLSE[row_base + r] = M + __logf(S);
    }
}

// ---------------------------------------------------------------------------
// Kernel 3: streaming per-row top-10 of sim = tn @ tn^T with diagonal zeroed.
// Same tiling; A strip staged in LDS; per-row sorted top-10 lists in LDS with
// threshold-filtered insertion; per-wave lists merged by wave 0.
// ---------------------------------------------------------------------------
__global__ void sim_topk_kernel(const _Float16* __restrict__ TN,
                                float* __restrict__ topv,
                                int* __restrict__ topi,
                                int N, int D) {
    const int lane = threadIdx.x & 31;
    const int wave = threadIdx.x >> 5;
    const int hi = lane >> 4;
    const int row_base = blockIdx.x * 16;

    __shared__ __align__(16) _Float16 As[16 * DMAX];
    __shared__ float stage[8][16][17];   // padded to dodge bank conflicts
    __shared__ float bv[8][16][TOPK];
    __shared__ int   bi[8][16][TOPK];

    {
        const v8h* src = (const v8h*)(TN + (size_t)row_base * D);
        v8h* dst = (v8h*)As;
        const int cnt = 16 * D / 8;
        for (int idx = threadIdx.x; idx < cnt; idx += blockDim.x)
            dst[idx] = src[idx];
    }
    if (lane < 16) {
#pragma unroll
        for (int q = 0; q < TOPK; ++q) { bv[wave][lane][q] = -3.0e38f; bi[wave][lane][q] = -1; }
    }
    __syncthreads();

    const _Float16* pA = As + (lane & 15) * D;
    const int ntile = N / 16;

    for (int ct = wave; ct < ntile; ct += 8) {
        const _Float16* pB = TN + (size_t)(ct * 16 + (lane & 15)) * D;
        v8f c = tile_gemm_lds(pA, pB, D, hi);
#pragma unroll
        for (int v = 0; v < 8; ++v)
            stage[wave][v + 8 * hi][lane & 15] = c[v];
        __builtin_amdgcn_wave_barrier();   // same-wave DS is in-order
        if (lane < 16) {
            const int row = row_base + lane;
            float thr = bv[wave][lane][TOPK - 1];
            for (int n = 0; n < 16; ++n) {
                float v = stage[wave][lane][n];
                int col = ct * 16 + n;
                if (col == row) v = 0.0f;          // sim * (1 - eye)
                if (v > thr) {
                    int p = TOPK - 1;
                    while (p > 0 && bv[wave][lane][p - 1] < v) {
                        bv[wave][lane][p] = bv[wave][lane][p - 1];
                        bi[wave][lane][p] = bi[wave][lane][p - 1];
                        --p;
                    }
                    bv[wave][lane][p] = v;
                    bi[wave][lane][p] = col;
                    thr = bv[wave][lane][TOPK - 1];
                }
            }
        }
    }
    __syncthreads();
    if (threadIdx.x < 16) {   // wave 0 merges the 8 per-wave lists per row
        const int r = threadIdx.x;
        for (int w = 1; w < 8; ++w) {
            for (int q = 0; q < TOPK; ++q) {
                float v = bv[w][r][q];
                int cI = bi[w][r][q];
                if (v > bv[0][r][TOPK - 1]) {
                    int p = TOPK - 1;
                    while (p > 0 && bv[0][r][p - 1] < v) {
                        bv[0][r][p] = bv[0][r][p - 1];
                        bi[0][r][p] = bi[0][r][p - 1];
                        --p;
                    }
                    bv[0][r][p] = v;
                    bi[0][r][p] = cI;
                }
            }
        }
        const int row = row_base + r;
#pragma unroll
        for (int q = 0; q < TOPK; ++q) {
            topv[row * TOPK + q] = bv[0][r][q];
            topi[row * TOPK + q] = bi[0][r][q];
        }
    }
}

// ---------------------------------------------------------------------------
// Kernel 4: per-row sparse CE contribution.
//   partial[i] = rowLSE_i + colLSE_i - scale * sum_j l_ij (img_i.txt_j + txt_i.img_j)
// with <=11 nonzero labels per row; dot-products in fp32 from original inputs.
// ---------------------------------------------------------------------------
__global__ void labels_ce_kernel(const float* __restrict__ img,
                                 const float* __restrict__ txt,
                                 const long long* __restrict__ cls,
                                 const float* __restrict__ topv,
                                 const int* __restrict__ topi,
                                 const float* __restrict__ rowLSE,
                                 const float* __restrict__ colLSE,
                                 const float* __restrict__ scale_p,
                                 float* __restrict__ partial,
                                 int N, int D) {
    const int i = blockIdx.x;
    __shared__ float l_s[TOPK + 1];
    __shared__ int   l_j[TOPK + 1];
    if (threadIdx.x == 0) {
        const long long ci = cls[i];
        float sv[TOPK + 1];
        int   sj[TOPK + 1];
        sv[0] = 1.0f;   // sim_ii ~ 1 (normalized); mask_sim/mask_class diag = 1
        sj[0] = i;
        float ssum = 1.0f;
        for (int q = 0; q < TOPK; ++q) {
            int j = topi[i * TOPK + q];
            float v = topv[i * TOPK + q];
            float mk = (j >= 0 && cls[j] == ci) ? 1.0f : 0.0f;
            v *= mk;
            sj[q + 1] = (j >= 0) ? j : i;
            sv[q + 1] = v;
            ssum += v;
        }
        const float inv = 1.0f / ssum;
        for (int q = 0; q <= TOPK; ++q) { l_s[q] = sv[q] * inv; l_j[q] = sj[q]; }
    }
    __syncthreads();

    const float* ii = img + (size_t)i * D;
    const float* ti = txt + (size_t)i * D;
    float acc = 0.f;
    for (int q = 0; q <= TOPK; ++q) {
        const float lq = l_s[q];       // uniform across block
        if (lq == 0.f) continue;
        const int j = l_j[q];
        const float* tj = txt + (size_t)j * D;
        const float* ij = img + (size_t)j * D;
        float d = 0.f;
        for (int d0 = threadIdx.x; d0 < D; d0 += blockDim.x)
            d += ii[d0] * tj[d0] + ti[d0] * ij[d0];
        acc += lq * d;
    }
    __shared__ float red[8];
#pragma unroll
    for (int off = 16; off > 0; off >>= 1) acc += __shfl_down(acc, off, 32);
    const int lane = threadIdx.x & 31, w = threadIdx.x >> 5;
    if (lane == 0) red[w] = acc;
    __syncthreads();
    if (threadIdx.x == 0) {
        float t = 0.f;
        for (int k = 0; k < 8; ++k) t += red[k];
        partial[i] = rowLSE[i] + colLSE[i] - (*scale_p) * t;
    }
}

// ---------------------------------------------------------------------------
// Kernel 5: deterministic fixed-order final reduction: out = sum(partial)/(2N)
// ---------------------------------------------------------------------------
__global__ void final_reduce_kernel(const float* __restrict__ partial,
                                    float* __restrict__ out, int N) {
    __shared__ float sm[256];
    float a = 0.f;
    for (int i = threadIdx.x; i < N; i += 256) a += partial[i];
    sm[threadIdx.x] = a;
    __syncthreads();
    for (int s = 128; s > 0; s >>= 1) {
        if (threadIdx.x < s) sm[threadIdx.x] += sm[threadIdx.x + s];
        __syncthreads();
    }
    if (threadIdx.x == 0) out[0] = sm[0] / (2.0f * (float)N);
}

// ---------------------------------------------------------------------------
extern "C" void kernel_launch(void* const* d_in, const int* in_sizes, int n_in,
                              void* d_out, int out_size, void* d_ws, size_t ws_size,
                              hipStream_t stream) {
    const float*     img   = (const float*)d_in[0];
    const float*     txt   = (const float*)d_in[1];
    const float*     scale = (const float*)d_in[2];
    const long long* cls   = (const long long*)d_in[3];

    const int N = in_sizes[3];
    const int D = in_sizes[0] / N;

    char* ws = (char*)d_ws;
    _Float16* img_h = (_Float16*)ws; ws += (size_t)N * D * sizeof(_Float16);
    _Float16* txt_h = (_Float16*)ws; ws += (size_t)N * D * sizeof(_Float16);
    _Float16* tn_h  = (_Float16*)ws; ws += (size_t)N * D * sizeof(_Float16);
    float* rowLSE   = (float*)ws;    ws += (size_t)N * sizeof(float);
    float* colLSE   = (float*)ws;    ws += (size_t)N * sizeof(float);
    float* topv     = (float*)ws;    ws += (size_t)N * TOPK * sizeof(float);
    int*   topi     = (int*)ws;      ws += (size_t)N * TOPK * sizeof(int);
    float* partial  = (float*)ws;    ws += (size_t)N * sizeof(float);

    prep_kernel<<<N, 256, 0, stream>>>(img, txt, img_h, txt_h, tn_h, N, D);
    gemm_lse_kernel<<<N / 16, 256, 0, stream>>>(img_h, txt_h, scale, rowLSE, N, D);
    gemm_lse_kernel<<<N / 16, 256, 0, stream>>>(txt_h, img_h, scale, colLSE, N, D);
    sim_topk_kernel<<<N / 16, 256, 0, stream>>>(tn_h, topv, topi, N, D);
    labels_ce_kernel<<<N, 256, 0, stream>>>(img, txt, cls, topv, topi,
                                            rowLSE, colLSE, scale, partial, N, D);
    final_reduce_kernel<<<1, 256, 0, stream>>>(partial, (float*)d_out, N);
}
